// TransformerLM_84834194031105
// MI455X (gfx1250) — matrix operand order repaired and checked by reading, not compiled, not run on hardware
//
#include <hip/hip_runtime.h>
#include <hip/hip_bf16.h>
#include <math.h>
#include <stdint.h>

// ---------------------------------------------------------------------------
// Sizes (compile-time, from the reference)
// ---------------------------------------------------------------------------
constexpr int Bc = 2, Sc = 2048, Dc = 1024, Hc = 16, Lc = 4, Fc = 2688, Vc = 32000;
constexpr int Mc = Bc * Sc;          // 4096 token rows
constexpr float EPSc = 1e-5f;

// ---------------------------------------------------------------------------
// CDNA5 WMMA types
// ---------------------------------------------------------------------------
typedef __attribute__((ext_vector_type(16))) __bf16 v16bf;
typedef __attribute__((ext_vector_type(8)))  __bf16 v8bf;
typedef __attribute__((ext_vector_type(8)))  float  v8f;

union V16 { v16bf v; v8bf h[2]; };

static __device__ __forceinline__ v8f vzero8() {
    v8f z = {0.f, 0.f, 0.f, 0.f, 0.f, 0.f, 0.f, 0.f};
    return z;
}

// ---------------------------------------------------------------------------
// Fragment builders (this file's contiguous k order, used identically for A and B):
//  A 16x32 bf16 : lane m=lane&15, half=lane>>4; a[0..15] <- K[half*16 .. +15]
//  B 32x16 bf16 : lane n=lane&15, half=lane>>4; b[0..15] <- K[half*16 .. +15]
//  C/D 16x16 f32: lane n=lane&15; vgpr r -> m = half*8 + r
// ---------------------------------------------------------------------------
static __device__ __forceinline__ v16bf load_a_frag(const __bf16* arow, int k0, int half) {
    V16 a;
    a.h[0] = *(const v8bf*)(arow + k0 + half * 16);
    a.h[1] = *(const v8bf*)(arow + k0 + half * 16 + 8);
    return a.v;
}

static __device__ __forceinline__ v16bf load_b_frag_bf16(const __bf16* row, int k0, int half) {
    const __bf16* p = row + k0 + half * 16;
    V16 b;
    b.h[0] = *(const v8bf*)(p);
    b.h[1] = *(const v8bf*)(p + 8);
    return b.v;
}

static __device__ __forceinline__ v16bf load_b_frag_f32(const float* __restrict__ wrow,
                                                        int k0, int half) {
    const float4* wp = (const float4*)(wrow + k0 + half * 16);
    float wt[16];
    *(float4*)(wt + 0)  = wp[0];
    *(float4*)(wt + 4)  = wp[1];
    *(float4*)(wt + 8)  = wp[2];
    *(float4*)(wt + 12) = wp[3];
    V16 b;
#pragma unroll
    for (int j = 0; j < 16; j++) b.v[j] = (__bf16)wt[j];
    return b.v;
}

#define WMMA_BF16(a, b, c) \
    __builtin_amdgcn_wmma_f32_16x16x32_bf16(false, (a), false, (b), (short)0, (c), false, false)

// gfx1250 async global -> LDS copy (16B per lane); INST_OFFSET applies to both
// the LDS and the global address, so one VGPR pair serves multiple chunks.
static __device__ __forceinline__ void async_copy_b128(uint32_t lds_byte, const void* gaddr) {
    asm volatile("global_load_async_to_lds_b128 %0, %1, off"
                 :: "v"(lds_byte), "v"((uint64_t)(uintptr_t)gaddr) : "memory");
}
static __device__ __forceinline__ void async_copy_b128_off64(uint32_t lds_byte, const void* gaddr) {
    asm volatile("global_load_async_to_lds_b128 %0, %1, off offset:64"
                 :: "v"(lds_byte), "v"((uint64_t)(uintptr_t)gaddr) : "memory");
}
static __device__ __forceinline__ void wait_async0() {
    asm volatile("s_wait_asynccnt 0x0" ::: "memory");
}

// ---------------------------------------------------------------------------
// Embedding gather: x[b,s,:] = tok_emb[idx[b,s],:]
// ---------------------------------------------------------------------------
__global__ __launch_bounds__(256) void k_embed(const int* __restrict__ idx,
                                               const float* __restrict__ emb,
                                               float* __restrict__ x, int n) {
    int i = blockIdx.x * 256 + threadIdx.x;
    if (i >= n) return;
    int row = i >> 10;          // / Dc (=1024)
    int col = i & (Dc - 1);
    x[i] = emb[(size_t)idx[row] * Dc + col];
}

// ---------------------------------------------------------------------------
// RMSNorm (fp32 in) -> bf16 out, fused gamma
// ---------------------------------------------------------------------------
__global__ __launch_bounds__(256) void k_rmsnorm(const float* __restrict__ x,
                                                 const float* __restrict__ g,
                                                 __bf16* __restrict__ out) {
    __shared__ float red[256];
    int row = blockIdx.x, tid = threadIdx.x;
    const float* xr = x + (size_t)row * Dc;
    float s = 0.f;
#pragma unroll
    for (int j = 0; j < Dc / 256; j++) { float v = xr[tid + j * 256]; s += v * v; }
    red[tid] = s; __syncthreads();
    for (int t = 128; t > 0; t >>= 1) { if (tid < t) red[tid] += red[tid + t]; __syncthreads(); }
    float rs = rsqrtf(red[0] * (1.0f / Dc) + EPSc);
    __bf16* orow = out + (size_t)row * Dc;
#pragma unroll
    for (int j = 0; j < Dc / 256; j++) {
        int c = tid + j * 256;
        orow[c] = (__bf16)(xr[c] * rs * g[c]);
    }
}

// ---------------------------------------------------------------------------
// GEMM: C[M,N] = A[M,K](bf16) @ W[N,K]^T (fp32 weights, cast in-register).
// Each wave computes a 16x32 tile (A-fragment reused across 2 WMMAs/K-step).
// ---------------------------------------------------------------------------
template <bool ADD_RESID, bool BF16_OUT>
__global__ __launch_bounds__(128) void k_gemm_t(const __bf16* __restrict__ A,
                                                const float* __restrict__ W,
                                                const float* __restrict__ resid,
                                                void* __restrict__ Cv,
                                                int M, int N, int K) {
    const int lane = threadIdx.x & 31;
    const int wave = threadIdx.x >> 5;
    const int n0 = blockIdx.x * 32;
    const int m0 = blockIdx.y * 64 + wave * 16;
    const int lm = lane & 15, half = lane >> 4;

    const __bf16* arow  = A + (size_t)(m0 + lm) * K;
    const float*  wrow0 = W + (size_t)(n0 + lm) * K;
    const float*  wrow1 = wrow0 + (size_t)16 * K;

    v8f acc0 = vzero8(), acc1 = vzero8();
    for (int k0 = 0; k0 < K; k0 += 32) {
        __builtin_prefetch(wrow0 + k0 + 128, 0, 0);   // global_prefetch_b8
        __builtin_prefetch(wrow1 + k0 + 128, 0, 0);
        v16bf a  = load_a_frag(arow, k0, half);
        v16bf b0 = load_b_frag_f32(wrow0, k0, half);
        v16bf b1 = load_b_frag_f32(wrow1, k0, half);
        acc0 = WMMA_BF16(a, b0, acc0);
        acc1 = WMMA_BF16(a, b1, acc1);
    }

    const int mbase = m0 + half * 8;
#pragma unroll
    for (int r = 0; r < 8; r++) {
        size_t rowoff = (size_t)(mbase + r) * N;
        size_t i0 = rowoff + n0 + lm;
        size_t i1 = rowoff + n0 + 16 + lm;
        if (BF16_OUT) {
            __bf16* C = (__bf16*)Cv;
            C[i0] = (__bf16)acc0[r];
            C[i1] = (__bf16)acc1[r];
        } else {
            float* C = (float*)Cv;
            float v0 = acc0[r], v1 = acc1[r];
            if (ADD_RESID) { v0 += resid[i0]; v1 += resid[i1]; }
            C[i0] = v0;
            C[i1] = v1;
        }
    }
}

// ---------------------------------------------------------------------------
// Fused SwiGLU up-projection: G[M,F] = bf16( silu(A@W1^T) * (A@W3^T) )
// ---------------------------------------------------------------------------
__global__ __launch_bounds__(128) void k_gemm_swiglu(const __bf16* __restrict__ A,
                                                     const float* __restrict__ W1,
                                                     const float* __restrict__ W3,
                                                     __bf16* __restrict__ G,
                                                     int M, int N, int K) {
    const int lane = threadIdx.x & 31;
    const int wave = threadIdx.x >> 5;
    const int n0 = blockIdx.x * 16;
    const int m0 = blockIdx.y * 64 + wave * 16;
    const int lm = lane & 15, half = lane >> 4;

    const __bf16* arow = A + (size_t)(m0 + lm) * K;
    const float*  w1r  = W1 + (size_t)(n0 + lm) * K;
    const float*  w3r  = W3 + (size_t)(n0 + lm) * K;

    v8f acc1 = vzero8(), acc3 = vzero8();
    for (int k0 = 0; k0 < K; k0 += 32) {
        __builtin_prefetch(w1r + k0 + 128, 0, 0);
        __builtin_prefetch(w3r + k0 + 128, 0, 0);
        v16bf a  = load_a_frag(arow, k0, half);
        v16bf b1 = load_b_frag_f32(w1r, k0, half);
        v16bf b3 = load_b_frag_f32(w3r, k0, half);
        acc1 = WMMA_BF16(a, b1, acc1);
        acc3 = WMMA_BF16(a, b3, acc3);
    }

    const int mbase = m0 + half * 8;
    const int cn = n0 + lm;
#pragma unroll
    for (int r = 0; r < 8; r++) {
        float g1 = acc1[r];
        // silu via v_rcp_f32 (precise IEEE division is a ~10-op expansion)
        float sil = g1 * __builtin_amdgcn_rcpf(1.f + __expf(-g1));
        G[(size_t)(mbase + r) * N + cn] = (__bf16)(sil * acc3[r]);
    }
}

// ---------------------------------------------------------------------------
// WMMA flash attention (causal). Block = 4 waves = 64 queries; wave = 16q x 64dh.
// Per 32-key step:
//   * K/V tiles [32 x 64] bf16 staged once per block via async global->LDS
//     (global_load_async_to_lds_b128 + s_wait_asynccnt, ASYNCcnt-tracked)
//   * S = Q K^T : 4 WMMAs (B-frags = contiguous ds reads from the K tile)
//   * online softmax, row stats via __shfl_xor over 16-lane N-groups
//   * P transposed C-layout -> A-layout through a per-wave LDS tile
//   * ctx += P V : 4 WMMAs (V B-frags = scalar ds reads from the V tile)
// Uniform trip count across the block (masked tiles contribute exactly zero),
// so __syncthreads around the shared staging is legal. Masking uses value
// selects only: EXEC stays all-ones for every WMMA.
// ---------------------------------------------------------------------------
__global__ __launch_bounds__(128) void k_attn(const __bf16* __restrict__ q,
                                              const __bf16* __restrict__ k,
                                              const __bf16* __restrict__ v,
                                              __bf16* __restrict__ ctx) {
    __shared__ __bf16 ktile[32 * 64];     // 4 KB, block-shared K tile
    __shared__ __bf16 vtile[32 * 64];     // 4 KB, block-shared V tile
    __shared__ __bf16 pbuf[4][16 * 32];   // per-wave P transpose tile

    const int tid  = threadIdx.x;
    const int lane = tid & 31, wave = tid >> 5;
    const int lk = lane & 15, half = lane >> 4;
    const int h = blockIdx.y, b = blockIdx.z;
    const int q0 = blockIdx.x * 64 + wave * 16;
    const int qlast = blockIdx.x * 64 + 63;       // last query row in block
    const size_t bS = (size_t)b * Sc;

    // staging assignment: thread -> (row, 16B chunk); 2 async copies per tensor
    const int srow = tid >> 2;                    // 0..31
    const int schunk = (tid & 3) * 16;            // byte 0/16/32/48 (+64 later)
    const uint32_t kdst = (uint32_t)(uintptr_t)&ktile[0] + srow * 128 + schunk;
    const uint32_t vdst = (uint32_t)(uintptr_t)&vtile[0] + srow * 128 + schunk;

    // Q fragments (A-layout), rows q0+lk, head slice [h*64, h*64+64)
    const __bf16* qrow = q + (bS + q0 + lk) * Dc + h * 64;
    const v16bf aq0 = load_a_frag(qrow, 0, half);
    const v16bf aq1 = load_a_frag(qrow, 32, half);

    float mrun[8], lrun[8];
    v8f acc[4];
#pragma unroll
    for (int r = 0; r < 8; r++) { mrun[r] = -3.0e38f; lrun[r] = 0.f; }
#pragma unroll
    for (int t = 0; t < 4; t++) acc[t] = vzero8();

    for (int kt = 0; kt <= qlast; kt += 32) {
        // ---- stage K/V tiles for the whole block (async -> LDS)
        const char* gk = (const char*)(k + (bS + kt + srow) * Dc + h * 64) + schunk;
        const char* gv = (const char*)(v + (bS + kt + srow) * Dc + h * 64) + schunk;
        async_copy_b128(kdst, gk);
        async_copy_b128_off64(kdst, gk);
        async_copy_b128(vdst, gv);
        async_copy_b128_off64(vdst, gv);
        wait_async0();
        __syncthreads();

        // ---- scores: S0 = keys [kt, kt+16), S1 = keys [kt+16, kt+32)
        const __bf16* kr0 = &ktile[(size_t)lk * 64];
        const __bf16* kr1 = &ktile[(size_t)(16 + lk) * 64];
        v8f s0 = vzero8(), s1 = vzero8();
        s0 = WMMA_BF16(aq0, load_b_frag_bf16(kr0, 0, half), s0);
        s0 = WMMA_BF16(aq1, load_b_frag_bf16(kr0, 32, half), s0);
        s1 = WMMA_BF16(aq0, load_b_frag_bf16(kr1, 0, half), s1);
        s1 = WMMA_BF16(aq1, load_b_frag_bf16(kr1, 32, half), s1);

        // ---- online softmax update (scale = 1/sqrt(64) = 0.125)
        const int key0 = kt + lk, key1 = kt + 16 + lk;
        float p0[8], p1[8];
#pragma unroll
        for (int r = 0; r < 8; r++) {
            const int qrw = q0 + half * 8 + r;
            float a0 = (key0 <= qrw) ? s0[r] * 0.125f : -3.0e38f;
            float a1 = (key1 <= qrw) ? s1[r] * 0.125f : -3.0e38f;
            float mx = fmaxf(a0, a1);
            mx = fmaxf(mx, __shfl_xor(mx, 1, 32));
            mx = fmaxf(mx, __shfl_xor(mx, 2, 32));
            mx = fmaxf(mx, __shfl_xor(mx, 4, 32));
            mx = fmaxf(mx, __shfl_xor(mx, 8, 32));
            const float mnew = fmaxf(mrun[r], mx);
            const float es = __expf(mrun[r] - mnew);
            mrun[r] = mnew;
            const float e0 = __expf(a0 - mnew);
            const float e1 = __expf(a1 - mnew);
            p0[r] = e0; p1[r] = e1;
            float ps = e0 + e1;
            ps += __shfl_xor(ps, 1, 32);
            ps += __shfl_xor(ps, 2, 32);
            ps += __shfl_xor(ps, 4, 32);
            ps += __shfl_xor(ps, 8, 32);
            lrun[r] = lrun[r] * es + ps;
#pragma unroll
            for (int t = 0; t < 4; t++) acc[t][r] *= es;
        }

        // ---- transpose P (C-layout -> A-layout) through per-wave LDS tile
#pragma unroll
        for (int r = 0; r < 8; r++) {
            pbuf[wave][(half * 8 + r) * 32 + lk]      = (__bf16)p0[r];
            pbuf[wave][(half * 8 + r) * 32 + 16 + lk] = (__bf16)p1[r];
        }
        __builtin_amdgcn_wave_barrier();           // LDS in-order per wave
        const __bf16* pr = &pbuf[wave][lk * 32];
        V16 pa;
        pa.h[0] = *(const v8bf*)(pr + half * 16);
        pa.h[1] = *(const v8bf*)(pr + half * 16 + 8);
        __builtin_amdgcn_wave_barrier();

        // ---- ctx accumulation: acc[t] += P(16x32) x V(32keys x 16dh)
#pragma unroll
        for (int t = 0; t < 4; t++) {
            V16 bv;
#pragma unroll
            for (int j = 0; j < 16; j++)
                bv.v[j] = vtile[(half * 16 + j) * 64 + t * 16 + lk];
            acc[t] = WMMA_BF16(pa.v, bv.v, acc[t]);
        }
        __syncthreads();                           // tiles fully consumed
    }

    // ---- normalize and write bf16 context
    const size_t obase = (bS + q0 + half * 8) * Dc + h * 64;
#pragma unroll
    for (int r = 0; r < 8; r++) {
        const float invl = __builtin_amdgcn_rcpf(lrun[r]);
        const size_t rb = obase + (size_t)r * Dc;
#pragma unroll
        for (int t = 0; t < 4; t++)
            ctx[rb + t * 16 + lk] = (__bf16)(acc[t][r] * invl);
    }
}

// ---------------------------------------------------------------------------
// Host orchestration
// ---------------------------------------------------------------------------
extern "C" void kernel_launch(void* const* d_in, const int* in_sizes, int n_in,
                              void* d_out, int out_size, void* d_ws, size_t ws_size,
                              hipStream_t stream) {
    (void)in_sizes; (void)n_in; (void)out_size; (void)ws_size;

    const int*   idx  = (const int*)  d_in[0];
    const float* temb = (const float*)d_in[1];
    const float* qw   = (const float*)d_in[2];
    const float* kw   = (const float*)d_in[3];
    const float* vw   = (const float*)d_in[4];
    const float* ow   = (const float*)d_in[5];
    const float* w1   = (const float*)d_in[6];
    const float* w2   = (const float*)d_in[7];
    const float* w3   = (const float*)d_in[8];
    const float* ln1  = (const float*)d_in[9];
    const float* ln2  = (const float*)d_in[10];
    const float* lnf  = (const float*)d_in[11];
    const float* lmh  = (const float*)d_in[12];
    float* out = (float*)d_out;

    // workspace carve-up (~77 MiB)
    char* wsp = (char*)d_ws;
    auto alloc = [&](size_t bytes) {
        void* p = (void*)wsp;
        wsp += (bytes + 255) & ~(size_t)255;
        return p;
    };
    float*  x    = (float*) alloc((size_t)Mc * Dc * sizeof(float));
    __bf16* hb   = (__bf16*)alloc((size_t)Mc * Dc * sizeof(__bf16));
    __bf16* qbf  = (__bf16*)alloc((size_t)Mc * Dc * sizeof(__bf16));
    __bf16* kbf  = (__bf16*)alloc((size_t)Mc * Dc * sizeof(__bf16));
    __bf16* vbf  = (__bf16*)alloc((size_t)Mc * Dc * sizeof(__bf16));
    __bf16* cbf  = (__bf16*)alloc((size_t)Mc * Dc * sizeof(__bf16));
    __bf16* gbf  = (__bf16*)alloc((size_t)Mc * Fc * sizeof(__bf16));

    const dim3 b256(256), b128(128);
    const dim3 gD(Dc / 32, Mc / 64);   // D-wide GEMMs (16x32 wave tiles)
    const dim3 gF(Fc / 16, Mc / 64);   // FFN up-proj (swiglu, 16-wide N)
    const dim3 gV(Vc / 32, Mc / 64);   // LM head

    k_embed<<<dim3((Mc * Dc + 255) / 256), b256, 0, stream>>>(idx, temb, x, Mc * Dc);

    for (int i = 0; i < Lc; i++) {
        const size_t wD = (size_t)i * Dc * Dc;
        const size_t wF = (size_t)i * Fc * Dc;

        // attention block
        k_rmsnorm<<<Mc, b256, 0, stream>>>(x, ln1 + (size_t)i * Dc, hb);
        k_gemm_t<false, true><<<gD, b128, 0, stream>>>(hb, qw + wD, nullptr, qbf, Mc, Dc, Dc);
        k_gemm_t<false, true><<<gD, b128, 0, stream>>>(hb, kw + wD, nullptr, kbf, Mc, Dc, Dc);
        k_gemm_t<false, true><<<gD, b128, 0, stream>>>(hb, vw + wD, nullptr, vbf, Mc, Dc, Dc);
        k_attn<<<dim3(Sc / 64, Hc, Bc), b128, 0, stream>>>(qbf, kbf, vbf, cbf);
        k_gemm_t<true, false><<<gD, b128, 0, stream>>>(cbf, ow + wD, x, x, Mc, Dc, Dc);

        // SwiGLU FFN block
        k_rmsnorm<<<Mc, b256, 0, stream>>>(x, ln2 + (size_t)i * Dc, hb);
        k_gemm_swiglu<<<gF, b128, 0, stream>>>(hb, w1 + wF, w3 + wF, gbf, Mc, Fc, Dc);
        k_gemm_t<true, false><<<gD, b128, 0, stream>>>(gbf, w2 + wF, x, x, Mc, Dc, Fc);
    }

    // final norm + LM head
    k_rmsnorm<<<Mc, b256, 0, stream>>>(x, lnf, hb);
    k_gemm_t<false, false><<<gV, b128, 0, stream>>>(hb, lmh, nullptr, out, Mc, Vc, Dc);
}
